// CustomHyperSemanticMessagePassing_58574763983241
// MI455X (gfx1250) — matrix operand order repaired and checked by reading, not compiled, not run on hardware
//
#include <hip/hip_runtime.h>
#include <hip/hip_bf16.h>

// Problem dims (fixed by the reference)
#define NN        32768
#define EE        16384
#define DD        4
#define KK        8
#define IN_DIMC   128
#define OUT_DIMC  128
#define EDGE_DIMC 64
#define NHEADS    8
#define HDIM      16
#define LKEYS     32   // DD*KK

typedef __attribute__((ext_vector_type(16))) _Float16 v16h;
typedef __attribute__((ext_vector_type(8)))  _Float16 v8h;
typedef __attribute__((ext_vector_type(4)))  _Float16 v4h;
typedef __attribute__((ext_vector_type(8)))  float    v8f;

// ---------------------------------------------------------------------------
// f32 -> f16 conversion
// ---------------------------------------------------------------------------
__global__ void f32_to_f16_kernel(const float* __restrict__ src,
                                  _Float16* __restrict__ dst, int n) {
  int i = blockIdx.x * blockDim.x + threadIdx.x;
  if (i < n) dst[i] = (_Float16)src[i];
}

// ---------------------------------------------------------------------------
// Compose weight matrices: Mq/Mk/Mv = W{q,k,v} @ W_lin  (each [128,128])
//                          Mke      = Wk @ W_edge        ([128,64])
// Tiny (7 MFLOP total) -> scalar fp32, store f16.
// ---------------------------------------------------------------------------
__global__ void compose_kernel(const float* __restrict__ Wlin,
                               const float* __restrict__ Wedge,
                               const float* __restrict__ Wq,
                               const float* __restrict__ Wk,
                               const float* __restrict__ Wv,
                               _Float16* __restrict__ Mq,
                               _Float16* __restrict__ Mk,
                               _Float16* __restrict__ Mv,
                               _Float16* __restrict__ Mke) {
  int t = blockIdx.x * blockDim.x + threadIdx.x;
  const int per = OUT_DIMC * IN_DIMC;
  if (t < 3 * per) {
    int m = t / per;
    int r = t % per;
    int o = r / IN_DIMC, i = r % IN_DIMC;
    const float* W = (m == 0) ? Wq : (m == 1) ? Wk : Wv;
    float acc = 0.f;
#pragma unroll 8
    for (int j = 0; j < OUT_DIMC; ++j)
      acc += W[o * OUT_DIMC + j] * Wlin[j * IN_DIMC + i];
    _Float16* M = (m == 0) ? Mq : (m == 1) ? Mk : Mv;
    M[o * IN_DIMC + i] = (_Float16)acc;
  } else {
    int r = t - 3 * per;
    if (r < OUT_DIMC * EDGE_DIMC) {
      int o = r / EDGE_DIMC, i = r % EDGE_DIMC;
      float acc = 0.f;
#pragma unroll 8
      for (int j = 0; j < OUT_DIMC; ++j)
        acc += Wk[o * OUT_DIMC + j] * Wedge[j * EDGE_DIMC + i];
      Mke[o * EDGE_DIMC + i] = (_Float16)acc;
    }
  }
}

// ---------------------------------------------------------------------------
// WMMA fragment helpers for v_wmma_f32_16x16x32_f16 (this code's contiguous k order, same for A and B):
//   A 16x32 f16: lane<16  holds row M=lane,    K k0..k0+15
//                lane>=16 holds row M=lane-16, K k0+16..k0+31
//   B 32x16 f16: lane holds column N=lane%16,  K = k0 + (lane/16)*16 .. +15
//   C 16x16 f32: VGPR r -> row = r + (lane/16)*8, col = lane%16
// B is fed row-major as B[col][k] (the projection matrix itself).
// ---------------------------------------------------------------------------
__device__ __forceinline__ v16h load_a_frag(const _Float16* __restrict__ A,
                                            int lda, int aRow, int k0,
                                            int lm, int hsel) {
  union { v16h v; v8h h[2]; } a;
  const _Float16* pa = A + (size_t)(aRow + lm) * lda + k0 + hsel * 16;
  a.h[0] = *(const v8h*)(pa);
  a.h[1] = *(const v8h*)(pa + 8);
  return a.v;
}

__device__ __forceinline__ v16h load_b_frag(const _Float16* __restrict__ B,
                                            int ldb, int bRow, int k0,
                                            int lm, int hsel) {
  return *(const v16h*)(B + (size_t)(bRow + lm) * ldb + k0 + hsel * 16);
}

// Generic 16x16 tile: out = A @ B^T + bias (bias must be non-null), opt. ReLU.
__device__ __forceinline__ void wmma_tile16x16(
    const _Float16* __restrict__ A, int lda, int aRow,
    const _Float16* __restrict__ B, int ldb, int bRow,
    int kTot, const float* __restrict__ bias,
    float* __restrict__ out, int ldo, bool relu) {
  const int lane = threadIdx.x & 31;
  const int hsel = lane >> 4;
  const int lm   = lane & 15;

  v8f c = {};
  for (int k0 = 0; k0 < kTot; k0 += 32) {
    v16h a = load_a_frag(A, lda, aRow, k0, lm, hsel);
    v16h b = load_b_frag(B, ldb, bRow, k0, lm, hsel);
    c = __builtin_amdgcn_wmma_f32_16x16x32_f16(
        false, a, false, b, (short)0, c, false, false);
  }

  const float bb = bias[bRow + lm];   // unconditional: no EXEC churn
#pragma unroll
  for (int r = 0; r < 8; ++r) {
    float v = c[r] + bb;
    if (relu) v = fmaxf(v, 0.f);
    out[(size_t)(aRow + r + hsel * 8) * ldo + (bRow + lm)] = v;
  }
}

// ---------------------------------------------------------------------------
// Fused node projections: Qn = xh@Mq^T + bq, Kn = xh@Mk^T, Vn = xh@Mv^T + bv
// One wave per 16-node tile; for each of 8 column tiles, a single shared A
// fragment per K-step feeds three WMMA accumulators (3x fewer A loads than
// running the projections as separate GEMMs).
// ---------------------------------------------------------------------------
__global__ void node_proj_kernel(const _Float16* __restrict__ xh,
                                 const _Float16* __restrict__ Mq,
                                 const _Float16* __restrict__ Mk,
                                 const _Float16* __restrict__ Mv,
                                 const float* __restrict__ bq,
                                 const float* __restrict__ bv,
                                 float* __restrict__ Qn,
                                 float* __restrict__ Kn,
                                 float* __restrict__ Vn) {
  const int wave = (blockIdx.x * blockDim.x + threadIdx.x) >> 5;
  if (wave >= NN / 16) return;
  const int aRow = wave * 16;
  const int lane = threadIdx.x & 31;
  const int hsel = lane >> 4;
  const int lm   = lane & 15;

#pragma unroll
  for (int ct = 0; ct < OUT_DIMC / 16; ++ct) {
    const int bRow = ct * 16;
    v8f cq = {}, ck = {}, cv = {};
#pragma unroll
    for (int k0 = 0; k0 < IN_DIMC; k0 += 32) {
      v16h a  = load_a_frag(xh, IN_DIMC, aRow, k0, lm, hsel);   // shared
      v16h b0 = load_b_frag(Mq, IN_DIMC, bRow, k0, lm, hsel);
      v16h b1 = load_b_frag(Mk, IN_DIMC, bRow, k0, lm, hsel);
      v16h b2 = load_b_frag(Mv, IN_DIMC, bRow, k0, lm, hsel);
      cq = __builtin_amdgcn_wmma_f32_16x16x32_f16(false, a, false, b0,
                                                  (short)0, cq, false, false);
      ck = __builtin_amdgcn_wmma_f32_16x16x32_f16(false, a, false, b1,
                                                  (short)0, ck, false, false);
      cv = __builtin_amdgcn_wmma_f32_16x16x32_f16(false, a, false, b2,
                                                  (short)0, cv, false, false);
    }
    const int   col = bRow + lm;
    const float biq = bq[col];
    const float biv = bv[col];
#pragma unroll
    for (int r = 0; r < 8; ++r) {
      const size_t row = (size_t)(aRow + r + hsel * 8) * OUT_DIMC + col;
      Qn[row] = cq[r] + biq;
      Kn[row] = ck[r];
      Vn[row] = cv[r] + biv;
    }
  }
}

// ---------------------------------------------------------------------------
// Edge projection: Ke = eh @ Mke^T + bk   [E,128], K = 64
// ---------------------------------------------------------------------------
__global__ void edge_proj_kernel(const _Float16* __restrict__ eh,
                                 const _Float16* __restrict__ Mke,
                                 const float* __restrict__ bk,
                                 float* __restrict__ Ke) {
  int wave = (blockIdx.x * blockDim.x + threadIdx.x) >> 5;
  if (wave >= EE / 16) return;
#pragma unroll
  for (int ct = 0; ct < OUT_DIMC / 16; ++ct)
    wmma_tile16x16(eh, EDGE_DIMC, wave * 16, Mke, EDGE_DIMC, ct * 16,
                   EDGE_DIMC, bk, Ke, OUT_DIMC, false);
}

// ---------------------------------------------------------------------------
// Attention: one wave32 per node. lane owns 4 consecutive dims
// (head h = lane/4, sub-dims (lane%4)*4..+3). L = 32 keys.
//   score[h,l] = q . (Kn[u]+Ke[e]) / sqrt(16) -> 4-lane shfl_xor reduce
//   softmax over l (lane-local, each lane keeps all 32 scores of its head)
//   ctx = sum_l attn[l] * Vn[u_l]   -> f16 workspace
// Gather tables (Kn/Vn/Ke ~40MB) are L2-resident on MI455X (192MB L2).
// ---------------------------------------------------------------------------
__global__ void attn_kernel(const float* __restrict__ Qn,
                            const float* __restrict__ Kn,
                            const float* __restrict__ Vn,
                            const float* __restrict__ Ke,
                            const int* __restrict__ node_edges,
                            const int* __restrict__ edge_nodes,
                            _Float16* __restrict__ ctxh) {
  const int lane = threadIdx.x & 31;
  const int n = (blockIdx.x * blockDim.x + threadIdx.x) >> 5;
  if (n >= NN) return;

  const int d0 = lane * 4;
  const float4 q = *(const float4*)(Qn + (size_t)n * OUT_DIMC + d0);

  int eidx[DD];
#pragma unroll
  for (int i = 0; i < DD; ++i) eidx[i] = node_edges[n * DD + i];

  int   u[LKEYS];
  float sc[LKEYS];
#pragma unroll
  for (int l = 0; l < LKEYS; ++l) {
    const int e  = eidx[l >> 3];
    const int uu = edge_nodes[e * KK + (l & 7)];
    u[l] = uu;
    const float4 kn = *(const float4*)(Kn + (size_t)uu * OUT_DIMC + d0);
    const float4 ke = *(const float4*)(Ke + (size_t)e  * OUT_DIMC + d0);
    __builtin_prefetch(Vn + (size_t)uu * OUT_DIMC + d0, 0, 3);  // global_prefetch
    float p = q.x * (kn.x + ke.x) + q.y * (kn.y + ke.y) +
              q.z * (kn.z + ke.z) + q.w * (kn.w + ke.w);
    // reduce the 4 lanes of this head (lanes 4h..4h+3)
    p += __shfl_xor(p, 1);
    p += __shfl_xor(p, 2);
    sc[l] = p * 0.25f;  // 1/sqrt(hd=16)
  }

  float mx = -3.402823e38f;
#pragma unroll
  for (int l = 0; l < LKEYS; ++l) mx = fmaxf(mx, sc[l]);
  float s = 0.f;
#pragma unroll
  for (int l = 0; l < LKEYS; ++l) { sc[l] = __expf(sc[l] - mx); s += sc[l]; }
  const float inv = 1.f / s;

  float ax = 0.f, ay = 0.f, az = 0.f, aw = 0.f;
#pragma unroll
  for (int l = 0; l < LKEYS; ++l) {
    const float w = sc[l] * inv;
    const float4 v = *(const float4*)(Vn + (size_t)u[l] * OUT_DIMC + d0);
    ax += w * v.x; ay += w * v.y; az += w * v.z; aw += w * v.w;
  }

  v4h h;
  h.x = (_Float16)ax; h.y = (_Float16)ay; h.z = (_Float16)az; h.w = (_Float16)aw;
  *(v4h*)(ctxh + (size_t)n * OUT_DIMC + d0) = h;
}

// ---------------------------------------------------------------------------
// Output projection: out = relu(ctx @ Wo^T + bo)  -- WMMA, K = 128
// ---------------------------------------------------------------------------
__global__ void out_proj_kernel(const _Float16* __restrict__ ctxh,
                                const _Float16* __restrict__ Woh,
                                const float* __restrict__ bo,
                                float* __restrict__ out) {
  int wave = (blockIdx.x * blockDim.x + threadIdx.x) >> 5;
  if (wave >= NN / 16) return;
#pragma unroll
  for (int ct = 0; ct < OUT_DIMC / 16; ++ct)
    wmma_tile16x16(ctxh, OUT_DIMC, wave * 16, Woh, OUT_DIMC, ct * 16,
                   OUT_DIMC, bo, out, OUT_DIMC, true);
}

// ---------------------------------------------------------------------------
extern "C" void kernel_launch(void* const* d_in, const int* in_sizes, int n_in,
                              void* d_out, int out_size, void* d_ws, size_t ws_size,
                              hipStream_t stream) {
  (void)in_sizes; (void)n_in; (void)out_size; (void)ws_size;

  const float* x         = (const float*)d_in[0];
  const float* edge_attr = (const float*)d_in[1];
  const int*   node_edges = (const int*)d_in[2];
  const int*   edge_nodes = (const int*)d_in[3];
  const float* W_lin  = (const float*)d_in[4];
  const float* W_edge = (const float*)d_in[5];
  const float* Wq = (const float*)d_in[6];
  const float* Wk = (const float*)d_in[7];
  const float* Wv = (const float*)d_in[8];
  const float* bq = (const float*)d_in[9];
  const float* bk = (const float*)d_in[10];
  const float* bv = (const float*)d_in[11];
  const float* Wo = (const float*)d_in[12];
  const float* bo = (const float*)d_in[13];
  float* out = (float*)d_out;

  // Workspace carve-up (~74 MB total)
  char* wp = (char*)d_ws;
  auto alloc = [&](size_t bytes) -> char* {
    char* p = wp;
    wp += (bytes + 255) & ~(size_t)255;
    return p;
  };
  _Float16* xh  = (_Float16*)alloc((size_t)NN * IN_DIMC * 2);
  _Float16* eh  = (_Float16*)alloc((size_t)EE * EDGE_DIMC * 2);
  _Float16* Woh = (_Float16*)alloc((size_t)OUT_DIMC * OUT_DIMC * 2);
  _Float16* Mq  = (_Float16*)alloc((size_t)OUT_DIMC * IN_DIMC * 2);
  _Float16* Mk  = (_Float16*)alloc((size_t)OUT_DIMC * IN_DIMC * 2);
  _Float16* Mv  = (_Float16*)alloc((size_t)OUT_DIMC * IN_DIMC * 2);
  _Float16* Mke = (_Float16*)alloc((size_t)OUT_DIMC * EDGE_DIMC * 2);
  float* Qn = (float*)alloc((size_t)NN * OUT_DIMC * 4);
  float* Kn = (float*)alloc((size_t)NN * OUT_DIMC * 4);
  float* Vn = (float*)alloc((size_t)NN * OUT_DIMC * 4);
  float* Ke = (float*)alloc((size_t)EE * OUT_DIMC * 4);
  _Float16* ctxh = (_Float16*)alloc((size_t)NN * OUT_DIMC * 2);

  // Stage 0: conversions
  {
    int n = NN * IN_DIMC;
    f32_to_f16_kernel<<<n / 256, 256, 0, stream>>>(x, xh, n);
  }
  {
    int n = EE * EDGE_DIMC;
    f32_to_f16_kernel<<<n / 256, 256, 0, stream>>>(edge_attr, eh, n);
  }
  {
    int n = OUT_DIMC * OUT_DIMC;
    f32_to_f16_kernel<<<n / 256, 256, 0, stream>>>(Wo, Woh, n);
  }

  // Stage 1: compose projection matrices (3*16384 + 8192 = 57344 threads)
  compose_kernel<<<(3 * OUT_DIMC * IN_DIMC + OUT_DIMC * EDGE_DIMC) / 256, 256, 0, stream>>>(
      W_lin, W_edge, Wq, Wk, Wv, Mq, Mk, Mv, Mke);

  // Stage 2: fused node projections (2048 waves -> 256 blocks of 8 waves)
  node_proj_kernel<<<(NN / 16) / 8, 256, 0, stream>>>(
      xh, Mq, Mk, Mv, bq, bv, Qn, Kn, Vn);

  // Stage 3: edge projection (1024 waves -> 128 blocks)
  edge_proj_kernel<<<(EE / 16) / 8, 256, 0, stream>>>(eh, Mke, bk, Ke);

  // Stage 4: attention (one wave per node -> 4096 blocks of 8 waves)
  attn_kernel<<<NN / 8, 256, 0, stream>>>(Qn, Kn, Vn, Ke,
                                          node_edges, edge_nodes, ctxh);

  // Stage 5: output projection + ReLU (2048 waves -> 256 blocks)
  out_proj_kernel<<<(NN / 16) / 8, 256, 0, stream>>>(ctxh, Woh, bo, out);
}
